// AWmix_41068477285109
// MI455X (gfx1250) — matrix operand order repaired and checked by reading, not compiled, not run on hardware
//
#include <hip/hip_runtime.h>
#include <hip/hip_bf16.h>

// ---------------------------------------------------------------------------
// Types for CDNA5 WMMA (wave32, 16x16x32 bf16, f32 accumulate)
// ---------------------------------------------------------------------------
typedef __bf16 v16bf __attribute__((ext_vector_type(16)));
typedef float  v8f   __attribute__((ext_vector_type(8)));

struct alignas(16) U4 { unsigned x, y, z, w; };
union FragCvt { U4 q[2]; v16bf v; };

__device__ __forceinline__ unsigned short f2bf(float f) {
    unsigned u = __builtin_bit_cast(unsigned, f);
    u += 0x7FFFu + ((u >> 16) & 1u);          // round-to-nearest-even
    return (unsigned short)(u >> 16);
}
__device__ __forceinline__ float bf2f(unsigned short h) {
    unsigned u = ((unsigned)h) << 16;
    return __builtin_bit_cast(float, u);
}

// A-fragment (16xK rows = spatial, lane L<16 holds K {0..7,16..23}, L>=16 holds {8..15,24..31})
__device__ __forceinline__ v16bf load_a_frag(const unsigned short* rowPtr, int k0, int sel) {
    FragCvt f;
    f.q[0] = *(const U4*)(rowPtr + k0 + sel * 8);
    f.q[1] = *(const U4*)(rowPtr + k0 + 16 + sel * 8);
    return f.v;
}
// B-fragment (Kx16 cols = out-channel m; lane L<16 holds K {0..7,16..23}, L>=16 holds {8..15,24..31})
__device__ __forceinline__ v16bf load_b_frag(const unsigned short* colPtr, int k0, int sel) {
    const U4* p = (const U4*)(colPtr + k0 + sel * 8);
    FragCvt f;
    f.q[0] = p[0];
    f.q[1] = p[2];
    return f.v;
}

#define WMMA_BF16(a, b, c) \
    __builtin_amdgcn_wmma_f32_16x16x32_bf16(false, (a), false, (b), (short)0, (c), false, false)

// ---------------------------------------------------------------------------
// Problem constants
// ---------------------------------------------------------------------------
static constexpr int Bn = 2;
static constexpr int C0 = 64;        // DIM
static constexpr int H0 = 256, W0 = 256, N0 = H0 * W0;   // 65536
static constexpr int H1 = 128, W1 = 128, N1 = H1 * W1;   // 16384
static constexpr int C1 = 256;       // 4*DIM

// ---------------------------------------------------------------------------
// Elementwise / layout kernels
// ---------------------------------------------------------------------------

// NCHW f32 -> NHWC bf16
__global__ void k_to_nhwc_bf16(const float* __restrict__ in, unsigned short* __restrict__ out,
                               int C, int N, int total) {
    int idx = blockIdx.x * blockDim.x + threadIdx.x;
    if (idx >= total) return;
    int n = idx % N;
    int c = (idx / N) % C;
    int b = idx / (N * C);
    out[((size_t)b * N + n) * C + c] = f2bf(in[idx]);
}

// flat f32 -> bf16
__global__ void k_f32_to_bf16(const float* __restrict__ in, unsigned short* __restrict__ out, int total) {
    int idx = blockIdx.x * blockDim.x + threadIdx.x;
    if (idx >= total) return;
    out[idx] = f2bf(in[idx]);
}

// conv weight OIHW (M,C,3,3) f32 -> [tap][M][C] bf16
__global__ void k_reorder_conv_w(const float* __restrict__ in, unsigned short* __restrict__ out,
                                 int M, int C, int total) {
    int idx = blockIdx.x * blockDim.x + threadIdx.x;
    if (idx >= total) return;
    int t = idx % 9;
    int c = (idx / 9) % C;
    int m = idx / (9 * C);
    out[((size_t)t * M + m) * C + c] = f2bf(in[idx]);
}

// Haar DWT: x NCHW f32 (B,64,256,256) -> NHWC bf16 (B, 16384, 256) channels [lh|hl|hh|ll]
__global__ void k_dwt(const float* __restrict__ x, unsigned short* __restrict__ xdp, int total) {
    int idx = blockIdx.x * blockDim.x + threadIdx.x;
    if (idx >= total) return;
    int p  = idx % N1;              // h2*128+w2
    int c  = (idx / N1) % C0;
    int b  = idx / (N1 * C0);
    int h2 = p / W1, w2 = p % W1;
    const float* xr = x + ((size_t)(b * C0 + c)) * N0;
    int h = 2 * h2, w = 2 * w2;
    float a  = xr[(size_t)h * W0 + w];
    float bb = xr[(size_t)h * W0 + w + 1];
    float cc = xr[(size_t)(h + 1) * W0 + w];
    float dd = xr[(size_t)(h + 1) * W0 + w + 1];
    float ll = (a + bb + cc + dd) * 0.5f;
    float lh = (a - bb + cc - dd) * 0.5f;
    float hl = (a + bb - cc - dd) * 0.5f;
    float hh = (a - bb - cc + dd) * 0.5f;
    unsigned short* o = xdp + ((size_t)b * N1 + p) * C1;
    o[c]           = f2bf(lh);
    o[64 + c]      = f2bf(hl);
    o[128 + c]     = f2bf(hh);
    o[192 + c]     = f2bf(ll);
}

// Haar IDWT: x_dwt NHWC bf16 (B,16384,256) (channel 4c+j, j=0:ll 1:lh 2:hl 3:hh)
//   -> x_idwt_pre NHWC bf16 (B, 65536, 64)
__global__ void k_idwt(const unsigned short* __restrict__ xd, unsigned short* __restrict__ xip, int total) {
    int idx = blockIdx.x * blockDim.x + threadIdx.x;
    if (idx >= total) return;
    int p = idx % N1;
    int c = (idx / N1) % C0;
    int b = idx / (N1 * C0);
    int h2 = p / W1, w2 = p % W1;
    const unsigned short* r = xd + ((size_t)b * N1 + p) * C1 + 4 * c;
    float yl = bf2f(r[0]), lh = bf2f(r[1]), hl = bf2f(r[2]), hh = bf2f(r[3]);
    float a  = (yl + lh + hl + hh) * 0.5f;
    float bb = (yl - lh + hl - hh) * 0.5f;
    float cc = (yl + lh - hl - hh) * 0.5f;
    float dd = (yl - lh - hl + hh) * 0.5f;
    int h = 2 * h2, w = 2 * w2;
    unsigned short* o = xip + (size_t)b * N0 * C0;
    o[((size_t)h * W0 + w) * C0 + c]           = f2bf(a);
    o[((size_t)h * W0 + w + 1) * C0 + c]       = f2bf(bb);
    o[((size_t)(h + 1) * W0 + w) * C0 + c]     = f2bf(cc);
    o[((size_t)(h + 1) * W0 + w + 1) * C0 + c] = f2bf(dd);
}

// ---------------------------------------------------------------------------
// WMMA GEMM for 1x1 convs:  Out[b][m][n'] = sum_k W[m][k] * X[b][n][k] + bias[m]
// X: NHWC bf16 [b][Nsp][K]; Wb: [M][K] bf16; Out: f32, with channel-fold reshape
// (fold=4 implements NCHW reshape (b,256,128,128)->(b,64,256,256) on the fly).
// One wave computes a 16(n) x 16(m) tile.
// ---------------------------------------------------------------------------
__global__ void k_gemm_wmma(const unsigned short* __restrict__ X,
                            const unsigned short* __restrict__ Wb,
                            const float* __restrict__ bias,
                            float* __restrict__ Out,
                            int K, int M, int Nsp, int log2fold, int nB) {
    int lane = threadIdx.x & 31;
    int sel  = lane >> 4;
    int l15  = lane & 15;
    int wave = blockIdx.x * (blockDim.x >> 5) + (threadIdx.x >> 5);
    int mt = M >> 4, nt = Nsp >> 4;
    int perB = mt * nt;
    int b = wave / perB;
    if (b >= nB) return;                    // wave-uniform guard
    int rem   = wave - b * perB;
    int mBase = (rem / nt) << 4;
    int nBase = (rem % nt) << 4;

    const unsigned short* rowPtr = X + ((size_t)b * Nsp + nBase + l15) * K;
    const unsigned short* colPtr = Wb + (size_t)(mBase + l15) * K;

    v8f acc = {0.f, 0.f, 0.f, 0.f, 0.f, 0.f, 0.f, 0.f};
    for (int k0 = 0; k0 < K; k0 += 32) {
        v16bf a = load_a_frag(rowPtr, k0, sel);
        v16bf w = load_b_frag(colPtr, k0, sel);
        acc = WMMA_BF16(a, w, acc);
    }
    int m = mBase + l15;
    if (bias) {
        float bv = bias[m];
#pragma unroll
        for (int r = 0; r < 8; ++r) acc[r] += bv;
    }
    int fold = 1 << log2fold;
    size_t outBase = (size_t)b * M * Nsp
                   + (size_t)(m >> log2fold) * ((size_t)fold * Nsp)
                   + (size_t)(m & (fold - 1)) * Nsp
                   + (size_t)nBase + sel * 8;   // 8 consecutive n per lane
    *(float4*)(Out + outBase)     = make_float4(acc[0], acc[1], acc[2], acc[3]);
    *(float4*)(Out + outBase + 4) = make_float4(acc[4], acc[5], acc[6], acc[7]);
}

// ---------------------------------------------------------------------------
// WMMA implicit-GEMM 3x3 conv, reflect pad 1, + bias + leaky_relu(0.01), bf16 out.
// X: NHWC bf16 [b][H][W][C]; Wt: [9][M][C] bf16; Out: bf16 NHWC with row stride
// outStride halves and channel offset chanOff (lets idwt-conv write into concat).
// ---------------------------------------------------------------------------
__global__ void k_conv3x3_wmma(const unsigned short* __restrict__ X,
                               const unsigned short* __restrict__ Wt,
                               const float* __restrict__ bias,
                               unsigned short* __restrict__ Out,
                               int C, int M, int H, int W,
                               int outStride, int chanOff, int nB) {
    int lane = threadIdx.x & 31;
    int sel  = lane >> 4;
    int l15  = lane & 15;
    int wave = blockIdx.x * (blockDim.x >> 5) + (threadIdx.x >> 5);
    int mt = M >> 4, wt = W >> 4;
    int perB = mt * H * wt;
    int b = wave / perB;
    if (b >= nB) return;                    // wave-uniform guard
    int rem   = wave - b * perB;
    int mBase = (rem / (H * wt)) << 4;
    int rem2  = rem % (H * wt);
    int h     = rem2 / wt;
    int wBase = (rem2 % wt) << 4;
    int w = wBase + l15;                    // this lane's A-row pixel column

    const unsigned short* Xb = X + (size_t)b * H * W * C;
    const unsigned short* colBase = Wt + (size_t)(mBase + l15) * C;

    v8f acc = {0.f, 0.f, 0.f, 0.f, 0.f, 0.f, 0.f, 0.f};
#pragma unroll
    for (int t = 0; t < 9; ++t) {
        int dy = t / 3 - 1, dx = t % 3 - 1;
        int hh = h + dy; hh = hh < 0 ? -hh : (hh >= H ? 2 * H - 2 - hh : hh);
        int ww = w + dx; ww = ww < 0 ? -ww : (ww >= W ? 2 * W - 2 - ww : ww);
        const unsigned short* rowPtr = Xb + ((size_t)hh * W + ww) * C;
        const unsigned short* colPtr = colBase + (size_t)t * M * C;
        for (int k0 = 0; k0 < C; k0 += 32) {
            v16bf a  = load_a_frag(rowPtr, k0, sel);
            v16bf wv = load_b_frag(colPtr, k0, sel);
            acc = WMMA_BF16(a, wv, acc);
        }
    }
    int m  = mBase + l15;
    float bv = bias[m];
    size_t nRow = (size_t)b * H * W + (size_t)h * W + wBase + sel * 8;
#pragma unroll
    for (int r = 0; r < 8; ++r) {
        float y = acc[r] + bv;
        y = y > 0.f ? y : 0.01f * y;        // leaky_relu(0.01)
        Out[(nRow + r) * outStride + chanOff + m] = f2bf(y);
    }
}

// ---------------------------------------------------------------------------
// Attention pieces (fp32, bandwidth-bound)
// ---------------------------------------------------------------------------

// inverse L2 norms of the 65536-long rows of q and k ([b][64][65536])
__global__ void k_rownorm(const float* __restrict__ q, const float* __restrict__ k,
                          float* __restrict__ qInv, float* __restrict__ kInv) {
    int r = blockIdx.x;               // 0..127 -> q rows, 128..255 -> k rows
    const float* src = (r < 128) ? q : k;
    float* dst       = (r < 128) ? qInv : kInv;
    int row = (r < 128) ? r : r - 128;
    const float* p = src + (size_t)row * N0;
    float s = 0.f;
    for (int i = threadIdx.x; i < N0; i += 256) { float v = p[i]; s += v * v; }
    __shared__ float red[256];
    red[threadIdx.x] = s;
    __syncthreads();
    for (int st = 128; st; st >>= 1) {
        if (threadIdx.x < st) red[threadIdx.x] += red[threadIdx.x + st];
        __syncthreads();
    }
    if (threadIdx.x == 0) dst[row] = 1.0f / fmaxf(sqrtf(red[0]), 1e-12f);
}

// Gram 16x16 per (b,head) with LDS tiling over n, then normalize*temp, softmax.
__global__ void k_gram_softmax(const float* __restrict__ q, const float* __restrict__ k,
                               const float* __restrict__ qInv, const float* __restrict__ kInv,
                               const float* __restrict__ temp, float* __restrict__ attn) {
    int bh = blockIdx.x;
    int b = bh >> 2, hd = bh & 3;
    __shared__ float qt[16 * 128];
    __shared__ float kt[16 * 128];
    __shared__ float sm[256];
    int t = threadIdx.x;
    int c = t >> 4, d = t & 15;
    const float* qb = q + ((size_t)b * C0 + hd * 16) * N0;
    const float* kb = k + ((size_t)b * C0 + hd * 16) * N0;
    float acc = 0.f;
    for (int n0 = 0; n0 < N0; n0 += 128) {
        __syncthreads();
#pragma unroll
        for (int i = 0; i < 8; ++i) {
            int idx = t + i * 256;
            int rr = idx >> 7, j = idx & 127;
            qt[idx] = qb[(size_t)rr * N0 + n0 + j];
            kt[idx] = kb[(size_t)rr * N0 + n0 + j];
        }
        __syncthreads();
        const float* qr = qt + c * 128;
        const float* kr = kt + d * 128;
        for (int j = 0; j < 128; ++j) acc = fmaf(qr[j], kr[j], acc);
    }
    float s = acc * qInv[b * C0 + hd * 16 + c] * kInv[b * C0 + hd * 16 + d] * temp[hd];
    __syncthreads();
    sm[t] = s;
    __syncthreads();
    if (t < 16) {                           // row-wise softmax over d
        float mx = -3.4e38f;
        for (int j = 0; j < 16; ++j) mx = fmaxf(mx, sm[t * 16 + j]);
        float e[16], sum = 0.f;
        for (int j = 0; j < 16; ++j) { e[j] = __expf(sm[t * 16 + j] - mx); sum += e[j]; }
        float inv = 1.f / sum;
        for (int j = 0; j < 16; ++j) attn[(size_t)bh * 256 + t * 16 + j] = e[j] * inv;
    }
}

// out_att[c][n] = sum_d attn[c][d]*v[d][n]; writes bf16 into concat[b][n][0..63]
__global__ void k_apply_attn(const float* __restrict__ v, const float* __restrict__ attn,
                             unsigned short* __restrict__ concat) {
    int blk = blockIdx.x;                   // 2*4*64 blocks
    int chunk = blk & 63;
    int bh = blk >> 6;
    int b = bh >> 2, hd = bh & 3;
    __shared__ float am[256];
    am[threadIdx.x] = attn[(size_t)bh * 256 + threadIdx.x];
    __syncthreads();
    int n = chunk * 1024 + threadIdx.x * 4;
    const float* vb = v + ((size_t)b * C0 + hd * 16) * N0 + n;
    float acc[16][4];
#pragma unroll
    for (int c = 0; c < 16; ++c)
        for (int j = 0; j < 4; ++j) acc[c][j] = 0.f;
#pragma unroll
    for (int d = 0; d < 16; ++d) {
        float4 vv = *(const float4*)(vb + (size_t)d * N0);
        float vr[4] = {vv.x, vv.y, vv.z, vv.w};
#pragma unroll
        for (int c = 0; c < 16; ++c) {
            float a = am[c * 16 + d];
            for (int j = 0; j < 4; ++j) acc[c][j] = fmaf(a, vr[j], acc[c][j]);
        }
    }
#pragma unroll
    for (int i = 0; i < 4; ++i) {
        unsigned short* o = concat + ((size_t)b * N0 + n + i) * 128 + hd * 16;
        for (int c = 0; c < 16; ++c) o[c] = f2bf(acc[c][i]);
    }
}

// ---------------------------------------------------------------------------
// Host orchestration
// ---------------------------------------------------------------------------
static inline size_t alignUp(size_t v) { return (v + 255) & ~(size_t)255; }

extern "C" void kernel_launch(void* const* d_in, const int* in_sizes, int n_in,
                              void* d_out, int out_size, void* d_ws, size_t ws_size,
                              hipStream_t stream) {
    (void)in_sizes; (void)n_in; (void)out_size; (void)ws_size;
    const float* x      = (const float*)d_in[0];
    const float* temp   = (const float*)d_in[1];
    const float* w1     = (const float*)d_in[2];
    const float* b1     = (const float*)d_in[3];
    const float* w2     = (const float*)d_in[4];
    const float* b2     = (const float*)d_in[5];
    const float* w3     = (const float*)d_in[6];
    const float* b3     = (const float*)d_in[7];
    const float* w_dwt  = (const float*)d_in[8];
    const float* b_dwt  = (const float*)d_in[9];
    const float* w_idwt = (const float*)d_in[10];
    const float* b_idwt = (const float*)d_in[11];
    const float* w_proj = (const float*)d_in[12];
    float* out = (float*)d_out;

    char* ws = (char*)d_ws;
    size_t off = 0;
    auto alloc = [&](size_t bytes) -> void* {
        void* p = ws + off;
        off = alignUp(off + bytes);
        return p;
    };

    unsigned short* xb     = (unsigned short*)alloc((size_t)Bn * N0 * C0 * 2);   // x NHWC bf16
    unsigned short* xdp    = (unsigned short*)alloc((size_t)Bn * N1 * C1 * 2);   // dwt concat NHWC bf16
    unsigned short* xd     = (unsigned short*)alloc((size_t)Bn * N1 * C1 * 2);   // after convd NHWC bf16
    unsigned short* xip    = (unsigned short*)alloc((size_t)Bn * N0 * C0 * 2);   // idwt NHWC bf16
    unsigned short* concat = (unsigned short*)alloc((size_t)Bn * N0 * 128 * 2);  // [out_att|x_idwt] NHWC bf16
    unsigned short* wdwtB  = (unsigned short*)alloc((size_t)9 * C1 * C1 * 2);
    unsigned short* widwtB = (unsigned short*)alloc((size_t)9 * C0 * C0 * 2);
    unsigned short* w1B    = (unsigned short*)alloc((size_t)C0 * C0 * 2);
    unsigned short* w2B    = (unsigned short*)alloc((size_t)C1 * C1 * 2);
    unsigned short* w3B    = (unsigned short*)alloc((size_t)C1 * C1 * 2);
    unsigned short* wprojB = (unsigned short*)alloc((size_t)C0 * 2 * C0 * 2);
    float* qf   = (float*)alloc((size_t)Bn * C0 * N0 * 4);                       // [b][64][65536]
    float* kf   = (float*)alloc((size_t)Bn * C0 * N0 * 4);
    float* vf   = (float*)alloc((size_t)Bn * C0 * N0 * 4);
    float* qInv = (float*)alloc((size_t)Bn * C0 * 4);
    float* kInv = (float*)alloc((size_t)Bn * C0 * 4);
    float* attn = (float*)alloc((size_t)Bn * 4 * 16 * 16 * 4);

    const int T = 256;
    auto blocks = [](long total, int t) { return (unsigned)((total + t - 1) / t); };

    // --- layout / weight conversion ---
    long txb = (long)Bn * C0 * N0;
    k_to_nhwc_bf16<<<blocks(txb, T), T, 0, stream>>>(x, xb, C0, N0, (int)txb);
    long tdwt = (long)Bn * C0 * N1;
    k_dwt<<<blocks(tdwt, T), T, 0, stream>>>(x, xdp, (int)tdwt);
    k_reorder_conv_w<<<blocks((long)C1 * C1 * 9, T), T, 0, stream>>>(w_dwt, wdwtB, C1, C1, C1 * C1 * 9);
    k_reorder_conv_w<<<blocks((long)C0 * C0 * 9, T), T, 0, stream>>>(w_idwt, widwtB, C0, C0, C0 * C0 * 9);
    k_f32_to_bf16<<<blocks(C0 * C0, T), T, 0, stream>>>(w1, w1B, C0 * C0);
    k_f32_to_bf16<<<blocks(C1 * C1, T), T, 0, stream>>>(w2, w2B, C1 * C1);
    k_f32_to_bf16<<<blocks(C1 * C1, T), T, 0, stream>>>(w3, w3B, C1 * C1);
    k_f32_to_bf16<<<blocks(C0 * 2 * C0, T), T, 0, stream>>>(w_proj, wprojB, C0 * 2 * C0);

    // --- 3x3 convd on dwt (256ch @128x128), WMMA ---
    {
        long waves = (long)Bn * (C1 / 16) * H1 * (W1 / 16);   // 32768
        k_conv3x3_wmma<<<(unsigned)(waves / 8), T, 0, stream>>>(
            xdp, wdwtB, b_dwt, xd, C1, C1, H1, W1, C1, 0, Bn);
    }
    // --- k, v, q 1x1 GEMMs (WMMA), outputs in attention layout ---
    {
        long wk = (long)Bn * (C1 / 16) * (N1 / 16);           // 32768
        k_gemm_wmma<<<(unsigned)(wk / 8), T, 0, stream>>>(xd, w2B, b2, kf, C1, C1, N1, 2, Bn);
        k_gemm_wmma<<<(unsigned)(wk / 8), T, 0, stream>>>(xd, w3B, b3, vf, C1, C1, N1, 2, Bn);
        long wq = (long)Bn * (C0 / 16) * (N0 / 16);           // 32768
        k_gemm_wmma<<<(unsigned)(wq / 8), T, 0, stream>>>(xb, w1B, b1, qf, C0, C0, N0, 0, Bn);
    }
    // --- IDWT + 3x3 convd (64ch @256x256) writing into concat channels [64,128) ---
    k_idwt<<<blocks(tdwt, T), T, 0, stream>>>(xd, xip, (int)tdwt);
    {
        long waves = (long)Bn * (C0 / 16) * H0 * (W0 / 16);   // 32768
        k_conv3x3_wmma<<<(unsigned)(waves / 8), T, 0, stream>>>(
            xip, widwtB, b_idwt, concat, C0, C0, H0, W0, 128, 64, Bn);
    }
    // --- attention: norms, gram+softmax, apply ---
    k_rownorm<<<256, T, 0, stream>>>(qf, kf, qInv, kInv);
    k_gram_softmax<<<Bn * 4, T, 0, stream>>>(qf, kf, qInv, kInv, temp, attn);
    k_apply_attn<<<Bn * 4 * 64, T, 0, stream>>>(vf, attn, concat);

    // --- final projection GEMM (no bias) -> d_out NCHW f32 ---
    {
        long wp = (long)Bn * (C0 / 16) * (N0 / 16);           // 32768
        k_gemm_wmma<<<(unsigned)(wp / 8), T, 0, stream>>>(concat, wprojB, nullptr, out, 2 * C0, C0, N0, 0, Bn);
    }
}
